// HybridQwen3LaCTBranch_48326972014918
// MI455X (gfx1250) — compile-verified, matrix operand order checked by reading, not run on hardware
//
#include <hip/hip_runtime.h>

typedef _Float16 v16h __attribute__((ext_vector_type(16)));
typedef _Float16 v4h  __attribute__((ext_vector_type(4)));
typedef float    v8f  __attribute__((ext_vector_type(8)));

#define BB 2
#define SS 4096
#define HIDD 2048
#define NQ_ 16
#define NKV_ 8
#define HD_ 128
#define NFW_ 4
#define FWD_ 512
#define DH_ 512
#define CHUNK_ 2048
#define NC_ 2
#define NBH 8        // B * NFW

// ---------------- GEMM (split-f16 WMMA, fp32 accumulate, double-buffered LDS) ----------------
// FORM_NT: C[M,N] = sum_k A[m,k] * B[n,k]     (A [M,K] k-contig, B [N,K] k-contig)
// FORM_NN: C[M,N] = sum_k A[m,k] * B[k,n]     (B n-contig)
// FORM_TN: C[M,N](+)= sum_c A[c,m]*rs[c] * B[c,n]   (contraction over rows)
constexpr int FORM_NT = 0;
constexpr int FORM_NN = 1;
constexpr int FORM_TN = 2;

#define BM 64
#define BN 32
#define BK 32
#define ASZ ((BM / 16) * 512)   // 2048 halves per A stage
#define BSZ ((BN / 16) * 512)   // 1024 halves per B stage

// LDS is laid out in WMMA *fragment order*: for each 16x32 subtile,
// value (row m, k) lives at [lane][elem] with
//   lane = (m & 15) + 16*((k >> 3) & 1)
//   elem = ((k >> 4) << 3) | (k & 7)
// so each lane's 16 fragment halves are 32 contiguous bytes -> one v16h load.
__device__ __forceinline__ int frag_off(int row, int k) {
    return ((row >> 4) << 9)                                  // 16-row subtile
         + ((((k >> 3) & 1) << 4) + (row & 15)) * 16          // lane * 16
         + (((k >> 4) << 3) + (k & 7));                       // element
}

template<int FORM>
__global__ __launch_bounds__(256)
void gemm_wmma(const float* __restrict__ A, long asb, long ash, int lda,
               const float* __restrict__ B, long bsb, long bsh, int ldb,
               float* __restrict__ C, long csb, long csh, int ldc,
               int M, int N, int K,
               const float* __restrict__ RS, long rssb, long rssh,
               int accFlag)
{
    __shared__ _Float16 AsH[2 * ASZ];
    __shared__ _Float16 AsL[2 * ASZ];
    __shared__ _Float16 BsH[2 * BSZ];
    __shared__ _Float16 BsL[2 * BSZ];

    const int bh = blockIdx.y;
    const int b  = bh >> 2;
    const int h  = bh & 3;
    A += (long)b * asb + (long)h * ash;
    B += (long)b * bsb + (long)h * bsh;
    C += (long)b * csb + (long)h * csh;
    const float* rs = (RS != nullptr) ? (RS + (long)b * rssb + (long)h * rssh) : nullptr;

    const int tilesN = N / BN;
    const int tm = blockIdx.x / tilesN;
    const int tn = blockIdx.x % tilesN;
    const int m0 = tm * BM;
    const int n0 = tn * BN;

    const int tid     = threadIdx.x;
    const int lane    = tid & 31;
    const int wave    = tid >> 5;
    const int wm      = wave & 3;   // 4 wave-rows  (4*16 = 64 = BM)
    const int wn      = wave >> 2;  // 2 wave-cols  (2*16 = 32 = BN)
    const int halfsel = lane >> 4;
    const int l15     = lane & 15;

    // ---- staging helpers (global -> hi/lo f16 in fragment-order LDS) ----
    auto stageA = [&](int k0, int sel) {
        _Float16* aH = AsH + sel * ASZ;
        _Float16* aL = AsL + sel * ASZ;
        if constexpr (FORM == FORM_NT || FORM == FORM_NN) {
            #pragma unroll
            for (int j = 0; j < 2; ++j) {
                int idx = tid + j * 256;            // 512 float4 loads
                int row = idx >> 3;
                int c4  = (idx & 7) << 2;
                const float* src = A + (long)(m0 + row) * lda + k0 + c4;
                __builtin_prefetch(src + BK, 0, 3); // speculative; safe past end
                float4 v = *reinterpret_cast<const float4*>(src);
                float vv[4] = {v.x, v.y, v.z, v.w};
                v4h hi, lo;
                #pragma unroll
                for (int q = 0; q < 4; ++q) {
                    hi[q] = (_Float16)vv[q];
                    lo[q] = (_Float16)(vv[q] - (float)hi[q]);
                }
                int off = frag_off(row, c4);        // multiple of 4 -> 8B aligned
                *reinterpret_cast<v4h*>(&aH[off]) = hi;
                *reinterpret_cast<v4h*>(&aL[off]) = lo;
            }
        } else { // FORM_TN: A global [c][m], transpose + optional per-row lr scale
            #pragma unroll
            for (int j = 0; j < 8; ++j) {
                int idx = tid + j * 256;            // 2048 scalars
                int kk = idx >> 6;
                int mm = idx & 63;
                float v = A[(long)(k0 + kk) * lda + m0 + mm];
                if (rs) v *= rs[(long)(k0 + kk) * 12];
                _Float16 hi = (_Float16)v;
                int off = frag_off(mm, kk);
                aH[off] = hi;
                aL[off] = (_Float16)(v - (float)hi);
            }
        }
    };
    auto stageB = [&](int k0, int sel) {
        _Float16* bHs = BsH + sel * BSZ;
        _Float16* bLs = BsL + sel * BSZ;
        if constexpr (FORM == FORM_NT) {
            int row = tid >> 3;
            int c4  = (tid & 7) << 2;
            const float* src = B + (long)(n0 + row) * ldb + k0 + c4;
            __builtin_prefetch(src + BK, 0, 3);
            float4 v = *reinterpret_cast<const float4*>(src);
            float vv[4] = {v.x, v.y, v.z, v.w};
            v4h hi, lo;
            #pragma unroll
            for (int q = 0; q < 4; ++q) {
                hi[q] = (_Float16)vv[q];
                lo[q] = (_Float16)(vv[q] - (float)hi[q]);
            }
            int off = frag_off(row, c4);
            *reinterpret_cast<v4h*>(&bHs[off]) = hi;
            *reinterpret_cast<v4h*>(&bLs[off]) = lo;
        } else { // NN / TN: B global [k][n] n-contig -> transpose into fragment order
            #pragma unroll
            for (int j = 0; j < 4; ++j) {
                int idx = tid + j * 256;            // 1024 scalars
                int kk = idx >> 5;
                int nn = idx & 31;
                float v = B[(long)(k0 + kk) * ldb + n0 + nn];
                _Float16 hi = (_Float16)v;
                int off = frag_off(nn, kk);
                bHs[off] = hi;
                bLs[off] = (_Float16)(v - (float)hi);
            }
        }
    };

    v8f acc = {0.f, 0.f, 0.f, 0.f, 0.f, 0.f, 0.f, 0.f};

    const int nt = K / BK;
    stageA(0, 0);
    stageB(0, 0);
    __syncthreads();

    #pragma unroll 2
    for (int kt = 0; kt < nt; ++kt) {
        const int cur = kt & 1;
        if (kt + 1 < nt) {                     // stage next tile into alternate buffer
            stageA((kt + 1) * BK, cur ^ 1);
            stageB((kt + 1) * BK, cur ^ 1);
        }
        // fragment loads: 32 contiguous bytes per lane (2x ds_load_b128 each)
        const int abase = cur * ASZ + (wm << 9) + (lane << 4);
        const int bbase = cur * BSZ + (wn << 9) + (lane << 4);
        const v16h aH = *reinterpret_cast<const v16h*>(&AsH[abase]);
        const v16h aL = *reinterpret_cast<const v16h*>(&AsL[abase]);
        const v16h bH = *reinterpret_cast<const v16h*>(&BsH[bbase]);
        const v16h bL = *reinterpret_cast<const v16h*>(&BsL[bbase]);

        // split-precision f32 emulation: hi*hi + lo*hi + hi*lo
        acc = __builtin_amdgcn_wmma_f32_16x16x32_f16(false, aH, false, bH, (short)0, acc, false, false);
        acc = __builtin_amdgcn_wmma_f32_16x16x32_f16(false, aL, false, bH, (short)0, acc, false, false);
        acc = __builtin_amdgcn_wmma_f32_16x16x32_f16(false, aH, false, bL, (short)0, acc, false, false);
        __syncthreads();   // next-stage writes visible / this-stage reads done before overwrite
    }

    // ---- write C tile: VGPR r -> row (halfsel*8 + r), col = lane&15 ----
    const int col = n0 + wn * 16 + l15;
    #pragma unroll
    for (int r = 0; r < 8; ++r) {
        int row = m0 + wm * 16 + halfsel * 8 + r;
        long off = (long)row * ldc + col;
        if (accFlag) C[off] += acc[r];
        else         C[off]  = acc[r];
    }
}

// ---------------- preprocess: GQA expand + scale/offset + rmsnorm+silu + rope ----------------
__global__ __launch_bounds__(256)
void preprocess(const float* __restrict__ fq, const float* __restrict__ fk, const float* __restrict__ fv,
                const float* __restrict__ qks, const float* __restrict__ qko,
                const float* __restrict__ qnw, const float* __restrict__ knw,
                const int* __restrict__ pid,
                float* __restrict__ qbuf, float* __restrict__ kbuf, float* __restrict__ vbuf)
{
    int blk = blockIdx.x;            // ((b*S + s)*NFW + h)
    int h  = blk & 3;
    int bs = blk >> 2;
    int b  = bs >> 12;               // S = 4096
    int s  = bs & 4095;
    int tid = threadIdx.x;
    __shared__ float red[256];

    long tok = (long)b * SS + s;
    float qv[2], kv[2], vv[2];
    #pragma unroll
    for (int u = 0; u < 2; ++u) {
        int j    = tid + u * 256;
        int cch  = h * FWD_ + j;     // flat channel in [0, 2048)
        int head = cch >> 7;
        int pos  = cch & 127;
        int kvh  = head >> 1;        // GQA group = 2
        float qr = fq[(tok * NQ_  + head) * HD_ + pos];
        float kr = fk[(tok * NKV_ + kvh ) * HD_ + pos];
        float vr = fv[(tok * NKV_ + kvh ) * HD_ + pos];
        qv[u] = qr * qks[2 * cch]     + qko[2 * cch];
        kv[u] = kr * qks[2 * cch + 1] + qko[2 * cch + 1];
        vv[u] = vr;
    }
    long obase = ((long)(b * NFW_ + h) * SS + s) * FWD_;
    vbuf[obase + tid]       = vv[0];
    vbuf[obase + tid + 256] = vv[1];

    // q: rmsnorm + silu
    red[tid] = qv[0] * qv[0] + qv[1] * qv[1];
    __syncthreads();
    for (int off = 128; off; off >>= 1) { if (tid < off) red[tid] += red[tid + off]; __syncthreads(); }
    float rq = rsqrtf(red[0] * (1.0f / 512.0f) + 1e-6f);
    __syncthreads();
    float q0 = qnw[tid]       * qv[0] * rq;
    float q1 = qnw[tid + 256] * qv[1] * rq;
    q0 = q0 / (1.f + expf(-q0));
    q1 = q1 / (1.f + expf(-q1));

    // k: rmsnorm + silu
    red[tid] = kv[0] * kv[0] + kv[1] * kv[1];
    __syncthreads();
    for (int off = 128; off; off >>= 1) { if (tid < off) red[tid] += red[tid + off]; __syncthreads(); }
    float rk = rsqrtf(red[0] * (1.0f / 512.0f) + 1e-6f);
    float k0 = knw[tid]       * kv[0] * rk;
    float k1 = knw[tid + 256] * kv[1] * rk;
    k0 = k0 / (1.f + expf(-k0));
    k1 = k1 / (1.f + expf(-k1));

    // rope: pair (j, j+256) shares freq index j
    float posf = (float)pid[tok];
    float invf = expf(-(float)tid * (13.8155106f / 256.0f));   // base 1e6, FWD 512
    float fr = posf * invf;
    float cs = cosf(fr), sn = sinf(fr);
    qbuf[obase + tid]       = q0 * cs - q1 * sn;
    qbuf[obase + tid + 256] = q1 * cs + q0 * sn;
    kbuf[obase + tid]       = k0 * cs - k1 * sn;
    kbuf[obase + tid + 256] = k1 * cs + k0 * sn;
}

// ---------------- per-token learning-rate projection ----------------
__global__ __launch_bounds__(256)
void lr_proj(const float* __restrict__ hid, const float* __restrict__ lrw,
             const float* __restrict__ lrb, float* __restrict__ lrB)
{
    int t = blockIdx.x;                       // b*S + s
    int tid = threadIdx.x;
    __shared__ float red[12 * 256];
    float p[12];
    #pragma unroll
    for (int l = 0; l < 12; ++l) p[l] = 0.f;
    const float* hx = hid + (long)t * HIDD;
    for (int j = 0; j < 8; ++j) {
        int ch = tid + j * 256;
        float x = hx[ch];
        #pragma unroll
        for (int l = 0; l < 12; ++l) p[l] += x * lrw[l * HIDD + ch];
    }
    #pragma unroll
    for (int l = 0; l < 12; ++l) red[l * 256 + tid] = p[l];
    __syncthreads();
    for (int off = 128; off; off >>= 1) {
        if (tid < off) {
            #pragma unroll
            for (int l = 0; l < 12; ++l) red[l * 256 + tid] += red[l * 256 + tid + off];
        }
        __syncthreads();
    }
    if (tid < 12) {
        float x = red[tid * 256] + lrb[tid] + (-6.9072554f);   // inv_softplus(1e-3)
        lrB[(long)t * 12 + tid] = (x > 20.f) ? x : log1pf(expf(x));
    }
}

// ---------------- fast-weight state init (broadcast over batch) ----------------
__global__ __launch_bounds__(256)
void init_w(const float* __restrict__ w0, const float* __restrict__ w1, const float* __restrict__ w2,
            float* __restrict__ W0, float* __restrict__ W1, float* __restrict__ W2)
{
    long i = (long)blockIdx.x * 256 + threadIdx.x;   // 8 * 512*512 elements
    long src = i & ((1L << 20) - 1);                 // mod (NFW*512*512)
    W0[i] = w0[src];
    W1[i] = w1[src];
    W2[i] = w2[src];
}

// ---------------- elementwise: x = silu(g)*h ----------------
__global__ __launch_bounds__(256)
void ew_apply(const float* __restrict__ G, const float* __restrict__ H, float* __restrict__ X)
{
    long i = (long)blockIdx.x * 256 + threadIdx.x;
    float g = G[i], hh = H[i];
    float sg = 1.f / (1.f + expf(-g));
    X[i] = g * sg * hh;
}

// ---------------- elementwise backward: hid, dgk, dhk ----------------
__global__ __launch_bounds__(256)
void ew_bwd(float* __restrict__ G, float* __restrict__ H,
            const float* __restrict__ D, float* __restrict__ X)
{
    long i = (long)blockIdx.x * 256 + threadIdx.x;
    float gk = G[i], hk = H[i], dh = D[i];
    float sg = 1.f / (1.f + expf(-gk));
    float silu = gk * sg;
    X[i] = silu * hk;                                  // hid
    G[i] = dh * hk * (sg * (1.f + gk * (1.f - sg)));   // dgk (unscaled; lr fused in TN gemm)
    H[i] = dh * silu;                                  // dhk
}

// ---------------- final rmsnorm over FWD per (b,s,h), in-place on d_out ----------------
__global__ __launch_bounds__(256)
void final_norm(float* __restrict__ out, const float* __restrict__ tnw)
{
    int blk = blockIdx.x;
    int h  = blk & 3;
    int bs = blk >> 2;
    int b  = bs >> 12;
    int s  = bs & 4095;
    int tid = threadIdx.x;
    __shared__ float red[256];
    long base = ((long)b * SS + s) * (NFW_ * FWD_) + (long)h * FWD_;
    float x0 = out[base + tid], x1 = out[base + tid + 256];
    red[tid] = x0 * x0 + x1 * x1;
    __syncthreads();
    for (int off = 128; off; off >>= 1) { if (tid < off) red[tid] += red[tid + off]; __syncthreads(); }
    float r = rsqrtf(red[0] * (1.0f / 512.0f) + 1e-6f);
    out[base + tid]       = tnw[tid]       * x0 * r;
    out[base + tid + 256] = tnw[tid + 256] * x1 * r;
}

// ---------------- host launch ----------------
extern "C" void kernel_launch(void* const* d_in, const int* in_sizes, int n_in,
                              void* d_out, int out_size, void* d_ws, size_t ws_size,
                              hipStream_t stream)
{
    const float* fq  = (const float*)d_in[0];
    const float* fk  = (const float*)d_in[1];
    const float* fv  = (const float*)d_in[2];
    const float* hs  = (const float*)d_in[3];
    const int*   pid = (const int*)  d_in[4];
    const float* w0  = (const float*)d_in[5];
    const float* w1  = (const float*)d_in[6];
    const float* w2  = (const float*)d_in[7];
    const float* lrw = (const float*)d_in[8];
    const float* lrb = (const float*)d_in[9];
    const float* qks = (const float*)d_in[10];
    const float* qko = (const float*)d_in[11];
    const float* qnw = (const float*)d_in[12];
    const float* knw = (const float*)d_in[13];
    const float* tnw = (const float*)d_in[14];
    float* out = (float*)d_out;

    float* ws = (float*)d_ws;
    const long QS  = (long)NBH * SS * FWD_;        // 16,777,216
    const long CS  = (long)NBH * CHUNK_ * FWD_;    //  8,388,608
    const long WSZ = (long)NBH * DH_ * FWD_;       //  2,097,152
    float* qbuf = ws;  ws += QS;
    float* kbuf = ws;  ws += QS;
    float* vbuf = ws;  ws += QS;
    float* bufG = ws;  ws += CS;
    float* bufH = ws;  ws += CS;
    float* bufX = ws;  ws += CS;
    float* bufD = ws;  ws += CS;
    float* W0s  = ws;  ws += WSZ;
    float* W1s  = ws;  ws += WSZ;
    float* W2s  = ws;  ws += WSZ;
    float* lrB  = ws;  ws += (long)BB * SS * 12;

    preprocess<<<BB * SS * NFW_, 256, 0, stream>>>(fq, fk, fv, qks, qko, qnw, knw, pid, qbuf, kbuf, vbuf);
    lr_proj<<<BB * SS, 256, 0, stream>>>(hs, lrw, lrb, lrB);
    init_w<<<(int)(WSZ / 256), 256, 0, stream>>>(w0, w1, w2, W0s, W1s, W2s);

    const long ASB = (long)NFW_ * SS * FWD_,     ASH = (long)SS * FWD_;       // q/k/v per-b / per-h strides
    const long CSB = (long)NFW_ * CHUNK_ * FWD_, CSH = (long)CHUNK_ * FWD_;   // chunk buffers
    const long WSB = (long)NFW_ * DH_ * FWD_,    WSH = (long)DH_ * FWD_;      // fast-weight state
    const dim3 gNT((CHUNK_ / BM) * (DH_ / BN), NBH);  // 32*16 tiles
    const dim3 gTN((DH_ / BM) * (FWD_ / BN), NBH);    //  8*16 tiles
    const int ewBlocks = (int)(CS / 256);

    for (int c = 0; c < NC_; ++c) {
        const float* qc = qbuf + (long)c * CHUNK_ * FWD_;
        const float* kc = kbuf + (long)c * CHUNK_ * FWD_;
        const float* vc = vbuf + (long)c * CHUNK_ * FWD_;
        const float* l0 = lrB + (long)c * CHUNK_ * 12 + 0;
        const float* l1 = lrB + (long)c * CHUNK_ * 12 + 4;
        const float* l2 = lrB + (long)c * CHUNK_ * 12 + 8;
        const long LRSB = (long)SS * 12;

        // apply (block-causal: uses W before this chunk's update)
        gemm_wmma<FORM_NT><<<gNT, 256, 0, stream>>>(qc, ASB, ASH, FWD_,  W0s, WSB, WSH, FWD_,
            bufG, CSB, CSH, FWD_, CHUNK_, DH_, FWD_, nullptr, 0, 0, 0);                 // gq
        gemm_wmma<FORM_NT><<<gNT, 256, 0, stream>>>(qc, ASB, ASH, FWD_,  W2s, WSB, WSH, FWD_,
            bufH, CSB, CSH, FWD_, CHUNK_, DH_, FWD_, nullptr, 0, 0, 0);                 // hq
        ew_apply<<<ewBlocks, 256, 0, stream>>>(bufG, bufH, bufX);                        // x = silu(gq)*hq
        gemm_wmma<FORM_NT><<<gNT, 256, 0, stream>>>(bufX, CSB, CSH, DH_, W1s, WSB, WSH, DH_,
            out + (long)c * CHUNK_ * (NFW_ * FWD_), (long)SS * (NFW_ * FWD_), (long)FWD_, NFW_ * FWD_,
            CHUNK_, FWD_, DH_, nullptr, 0, 0, 0);                                        // o (pre-norm)

        // update path
        gemm_wmma<FORM_NT><<<gNT, 256, 0, stream>>>(kc, ASB, ASH, FWD_,  W0s, WSB, WSH, FWD_,
            bufG, CSB, CSH, FWD_, CHUNK_, DH_, FWD_, nullptr, 0, 0, 0);                 // gk
        gemm_wmma<FORM_NT><<<gNT, 256, 0, stream>>>(kc, ASB, ASH, FWD_,  W2s, WSB, WSH, FWD_,
            bufH, CSB, CSH, FWD_, CHUNK_, DH_, FWD_, nullptr, 0, 0, 0);                 // hk
        gemm_wmma<FORM_NN><<<gNT, 256, 0, stream>>>(vc, ASB, ASH, FWD_,  W1s, WSB, WSH, DH_,
            bufD, CSB, CSH, DH_, CHUNK_, DH_, FWD_, nullptr, 0, 0, 0);                  // dhid = v @ W1
        ew_bwd<<<ewBlocks, 256, 0, stream>>>(bufG, bufH, bufD, bufX);                    // hid,dgk,dhk

        // weight updates: W += (rowscaled A)^T B
        gemm_wmma<FORM_TN><<<gTN, 256, 0, stream>>>(vc,   ASB, ASH, FWD_, bufX, CSB, CSH, FWD_,
            W1s, WSB, WSH, DH_,  FWD_, DH_,  CHUNK_, l1, LRSB, 1, 1);                   // W1 += (v*l1)^T hid
        gemm_wmma<FORM_TN><<<gTN, 256, 0, stream>>>(bufG, CSB, CSH, FWD_, kc,  ASB, ASH, FWD_,
            W0s, WSB, WSH, FWD_, DH_,  FWD_, CHUNK_, l0, LRSB, 1, 1);                   // W0 += (dgk*l0)^T k
        gemm_wmma<FORM_TN><<<gTN, 256, 0, stream>>>(bufH, CSB, CSH, FWD_, kc,  ASB, ASH, FWD_,
            W2s, WSB, WSH, FWD_, DH_,  FWD_, CHUNK_, l2, LRSB, 1, 1);                   // W2 += (dhk*l2)^T k
    }

    final_norm<<<BB * SS * NFW_, 256, 0, stream>>>(out, tnw);
}
